// SparseMotionEncoder_8607114461668
// MI455X (gfx1250) — matrix operand order repaired and checked by reading, not compiled, not run on hardware
//
#include <hip/hip_runtime.h>
#include <hip/hip_bf16.h>
#include <stdint.h>

// ---------------------------------------------------------------------------
// SparseMotionEncoder for gfx1250 (MI455X): bf16 WMMA gather-GEMM pipeline.
// All matmul-heavy layers run through v_wmma_f32_16x16x32_bf16 (f32 accum).
// Wave tile: 32 voxel rows x 64 output channels, ping-pong double-buffered
// loads so WMMAs overlap the L2 gather latency.
// ---------------------------------------------------------------------------

typedef __attribute__((ext_vector_type(8)))  float  v8f;
typedef __attribute__((ext_vector_type(16))) __bf16 v16bf;

// ---------------------------------------------------------------------------
// Weight packing: W[k][cin][cout] (f32) -> bf16 tiles of 32x16 (KxN), one tile
// = 512 bf16 = 1024B, laid out so that lane l reads its v16bf as 16 contiguous
// bf16 at tile*512 + l*16.  Assumed B-matrix VGPR layout (per ISA 7.12.2 /
// sparse-B analogy): lane l<16 holds col N=l, K=0..7 in V0-3, 16..23 in V4-7; lane l>=16
// holds col N=l-16, K=8..15 in V0-3, 24..31 in V4-7.
// ---------------------------------------------------------------------------
__global__ void pack_w_kernel(const float* __restrict__ W, __bf16* __restrict__ Wp,
                              int K, int Cin, int CinP, int Cout, int CoutP) {
  const int KC = CinP >> 5;
  const int NT = CoutP >> 4;
  const size_t total = (size_t)K * KC * NT * 512;
  size_t t = (size_t)blockIdx.x * blockDim.x + threadIdx.x;
  if (t >= total) return;
  int    e    = (int)(t & 511);
  size_t tile = t >> 9;
  int nt = (int)(tile % NT);
  int kc = (int)((tile / NT) % KC);
  int k  = (int)(tile / ((size_t)NT * KC));
  int l = e >> 4;            // lane 0..31
  int v = (e >> 1) & 7;      // vgpr 0..7
  int s = e & 1;             // low/high bf16 in dword
  int cin  = kc * 32 + ((l & 16) ? 8 : 0) + ((2 * v + s) & 7) + (((2 * v + s) & 8) ? 16 : 0);
  int cout = nt * 16 + (l & 15);
  float val = 0.0f;
  if (cin < Cin && cout < Cout)
    val = W[((size_t)k * Cin + cin) * Cout + cout];
  Wp[t] = (__bf16)val;
}

// corr_features f32 [N,324] -> bf16 [N,352] (K padded to multiple of 32)
__global__ void convert_corr_kernel(const float* __restrict__ src, __bf16* __restrict__ dst,
                                    int Nv) {
  size_t t = (size_t)blockIdx.x * blockDim.x + threadIdx.x;
  if (t >= (size_t)Nv * 352) return;
  int c = (int)(t % 352);
  int n = (int)(t / 352);
  dst[t] = (c < 324) ? (__bf16)src[(size_t)n * 324 + c] : (__bf16)0.0f;
}

__global__ void pad_bias_kernel(const float* __restrict__ b, float* __restrict__ bp,
                                int n, int np) {
  int i = threadIdx.x;
  if (i < np) bp[i] = (i < n) ? b[i] : 0.0f;
}

// cat[n, 0:192] = corr2 ; cat[n, 192:256] = f2   (bf16)
__global__ void concat_kernel(const __bf16* __restrict__ corr2, const __bf16* __restrict__ f2,
                              __bf16* __restrict__ cat, int Nv) {
  size_t t = (size_t)blockIdx.x * blockDim.x + threadIdx.x;
  if (t >= (size_t)Nv * 256) return;
  int c = (int)(t & 255);
  int n = (int)(t >> 8);
  cat[t] = (c < 192) ? corr2[(size_t)n * 192 + c] : f2[(size_t)n * 64 + (c - 192)];
}

// out[n, 120:122] = flow ; out[n, 122:128] = extrinsics   (f32)
__global__ void tail_kernel(const float* __restrict__ flow, const float* __restrict__ extr,
                            float* __restrict__ out, int Nv) {
  size_t t = (size_t)blockIdx.x * blockDim.x + threadIdx.x;
  if (t >= (size_t)Nv * 8) return;
  int c = (int)(t & 7);
  int n = (int)(t >> 3);
  out[(size_t)n * 128 + 120 + c] = (c < 2) ? flow[(size_t)n * 2 + c]
                                           : extr[(size_t)n * 6 + (c - 2)];
}

// ---------------------------------------------------------------------------
// flowconv1: k=343 taps, Cin=2, Cout=128.  VALU kernel; 16 voxels per block so
// each Wf1 row is reused 16x (Wf1 = 351KB, L2 resident).
// ---------------------------------------------------------------------------
__global__ __launch_bounds__(128)
void flow_conv1_kernel(const float* __restrict__ flow, const int* __restrict__ nbr7,
                       const float* __restrict__ Wf1, const float* __restrict__ bf1,
                       __bf16* __restrict__ out, int Nv) {
  __shared__ float sf0[16];
  __shared__ float sf1[16];
  const int co = threadIdx.x;             // output channel 0..127
  const int v0 = blockIdx.x * 16;         // first voxel of this block
  float acc[16];
#pragma unroll
  for (int v = 0; v < 16; ++v) acc[v] = 0.0f;
  const float b = bf1[co];
  for (int k = 0; k < 343; ++k) {
    if (co < 16) {
      int idx = nbr7[(size_t)k * Nv + v0 + co];
      float a0 = 0.0f, a1 = 0.0f;
      if (idx >= 0) { a0 = flow[(size_t)idx * 2]; a1 = flow[(size_t)idx * 2 + 1]; }
      sf0[co] = a0; sf1[co] = a1;
    }
    __syncthreads();
    const float w0 = Wf1[((size_t)k * 2 + 0) * 128 + co];
    const float w1 = Wf1[((size_t)k * 2 + 1) * 128 + co];
#pragma unroll
    for (int v = 0; v < 16; ++v)
      acc[v] = fmaf(sf0[v], w0, fmaf(sf1[v], w1, acc[v]));
    __syncthreads();
  }
#pragma unroll
  for (int v = 0; v < 16; ++v)
    out[(size_t)(v0 + v) * 128 + co] = (__bf16)fmaxf(acc[v] + b, 0.0f);
}

// ---------------------------------------------------------------------------
// Generic WMMA submanifold conv:
//   out[n,:] = relu( sum_k feat[nbr[k][n],:] @ W_k + bias )
// Wave tile: 32 rows (2 M-subtiles) x 64 cols (4 N-tiles) -> 8 accumulators.
// A-tile gather follows the documented 16-bit A layout:
//   lane l<16 : row m=l,   K = kbase+0..7  (V0-3) and kbase+16..23 (V4-7)
//   lane l>=16: row m=l-16, K = kbase+8..15 (V0-3) and kbase+24..31 (V4-7)
// Flattened (k,kc) loop with explicit ping-pong double buffering: loads for
// step i+1 are issued before the 8 WMMAs of step i.
// ---------------------------------------------------------------------------
struct ConvTile {
  uint4 a0[2];     // A subtile, rows m..m+15   (lo half, hi half)
  uint4 a1[2];     // A subtile, rows m+16..m+31
  uint4 b[4][2];   // 4 B tiles, 32B per lane each
};

__global__ __launch_bounds__(256)
void wmma_subm_conv_kernel(const __bf16* __restrict__ feat, int CinP,
                           const int* __restrict__ nbr, int K,
                           const __bf16* __restrict__ Wpack,
                           const float* __restrict__ bias,
                           __bf16* __restrict__ outB,
                           float* __restrict__ outF, int outFcols,
                           int CoutP, int Nv) {
  const int KC  = CinP >> 5;           // K-chunks of 32
  const int NT  = CoutP >> 4;          // 16-col tiles
  const int NG  = CoutP >> 6;          // groups of 4 tiles
  const int MT2 = (Nv + 31) >> 5;      // 32-row supertiles
  const int jobs = MT2 * NG;

  const int wave = threadIdx.x >> 5;
  const int lane = threadIdx.x & 31;
  const int job  = blockIdx.x * 8 + wave;
  if (job >= jobs) return;
  const int mt = job % MT2;
  const int ng = job / MT2;

  const int halfsel = lane >> 4;       // 0: K-half 0, 1: K-half 1
  const int mlocal  = lane & 15;
  const int m0 = mt * 32 + mlocal;     // row of A-subtile 0 for this lane
  const int m1 = m0 + 16;              // row of A-subtile 1

  v8f acc[2][4];
#pragma unroll
  for (int mi = 0; mi < 2; ++mi)
#pragma unroll
    for (int j = 0; j < 4; ++j)
#pragma unroll
      for (int i = 0; i < 8; ++i) acc[mi][j][i] = 0.0f;

  float bj[4];
#pragma unroll
  for (int j = 0; j < 4; ++j) bj[j] = bias[ng * 64 + j * 16 + mlocal];

  const uint4 z4 = make_uint4(0u, 0u, 0u, 0u);

  // ---- gather-row state for the k-offset currently being *loaded* ----
  int lk = 0, lkc = 0;
  bool v0 = false, v1 = false;
  const __bf16* fb0 = feat;
  const __bf16* fb1 = feat;

  auto fetchRows = [&](int k) {
    int r0, r1;
    if (nbr) {
      r0 = (m0 < Nv) ? nbr[(size_t)k * Nv + m0] : -1;
      r1 = (m1 < Nv) ? nbr[(size_t)k * Nv + m1] : -1;
    } else {
      r0 = (m0 < Nv) ? m0 : -1;
      r1 = (m1 < Nv) ? m1 : -1;
    }
    v0 = (r0 >= 0);
    v1 = (r1 >= 0);
    fb0 = feat + (size_t)(v0 ? r0 : 0) * CinP + (halfsel << 3);
    fb1 = feat + (size_t)(v1 ? r1 : 0) * CinP + (halfsel << 3);
  };

  auto loadTile = [&](ConvTile& T) {
    const __bf16* f0 = fb0 + lkc * 32;
    const __bf16* f1 = fb1 + lkc * 32;
    T.a0[0] = v0 ? *(const uint4*)(f0)      : z4;
    T.a0[1] = v0 ? *(const uint4*)(f0 + 16) : z4;
    T.a1[0] = v1 ? *(const uint4*)(f1)      : z4;
    T.a1[1] = v1 ? *(const uint4*)(f1 + 16) : z4;
    const __bf16* wp = Wpack + (((size_t)lk * KC + lkc) * NT + (size_t)ng * 4) * 512
                             + (size_t)lane * 16;
#pragma unroll
    for (int j = 0; j < 4; ++j) {
      T.b[j][0] = *(const uint4*)(wp + j * 512);
      T.b[j][1] = *(const uint4*)(wp + j * 512 + 8);
    }
    // advance load cursor
    if (++lkc == KC) {
      lkc = 0;
      ++lk;
      if (lk < K) fetchRows(lk);
    }
  };

  auto wmmaTile = [&](const ConvTile& T) {
    union { uint4 u[2]; v16bf v; } A0, A1, B;
    A0.u[0] = T.a0[0]; A0.u[1] = T.a0[1];
    A1.u[0] = T.a1[0]; A1.u[1] = T.a1[1];
#pragma unroll
    for (int j = 0; j < 4; ++j) {
      B.u[0] = T.b[j][0]; B.u[1] = T.b[j][1];
      acc[0][j] = __builtin_amdgcn_wmma_f32_16x16x32_bf16(
          false, A0.v, false, B.v, (short)0, acc[0][j], false, false);
      acc[1][j] = __builtin_amdgcn_wmma_f32_16x16x32_bf16(
          false, A1.v, false, B.v, (short)0, acc[1][j], false, false);
    }
  };

  // ---- software-pipelined main loop over flattened (k, kc) ----
  const int kkTot = K * KC;
  ConvTile T0, T1;
  fetchRows(0);
  loadTile(T0);
  int kk = 1;
  for (; kk + 1 < kkTot; kk += 2) {
    loadTile(T1);
    wmmaTile(T0);
    loadTile(T0);
    wmmaTile(T1);
  }
  if (kk < kkTot) {           // one step left after T0
    loadTile(T1);
    wmmaTile(T0);
    wmmaTile(T1);
  } else {
    wmmaTile(T0);
  }

  // ---- epilogue: bias + relu, store bf16 feature map and/or f32 output ----
#pragma unroll
  for (int mi = 0; mi < 2; ++mi) {
    const int mbase = mt * 32 + mi * 16 + (halfsel << 3);
#pragma unroll
    for (int j = 0; j < 4; ++j) {
      const int c = ng * 64 + j * 16 + mlocal;
#pragma unroll
      for (int v = 0; v < 8; ++v) {
        const int m = mbase + v;
        if (m < Nv) {
          float val = fmaxf(acc[mi][j][v] + bj[j], 0.0f);
          if (outB) outB[(size_t)m * CoutP + c] = (__bf16)val;
          if (outF && c < outFcols) outF[(size_t)m * 128 + c] = val;
        }
      }
    }
  }
}

// ---------------------------------------------------------------------------
extern "C" void kernel_launch(void* const* d_in, const int* in_sizes, int n_in,
                              void* d_out, int out_size, void* d_ws, size_t ws_size,
                              hipStream_t stream) {
  const float* extr = (const float*)d_in[0];    // [N,6]
  const float* flow = (const float*)d_in[1];    // [N,2]
  const float* corr = (const float*)d_in[2];    // [N,324]
  const int*   nbr3 = (const int*)d_in[3];      // [27,N]
  const int*   nbr7 = (const int*)d_in[4];      // [343,N]
  const float* Wc1  = (const float*)d_in[5];    // [324,256]
  const float* bc1  = (const float*)d_in[6];
  const float* Wc2  = (const float*)d_in[7];    // [27,256,192]
  const float* bc2  = (const float*)d_in[8];
  const float* Wf1  = (const float*)d_in[9];    // [343,2,128]
  const float* bf1  = (const float*)d_in[10];
  const float* Wf2  = (const float*)d_in[11];   // [27,128,64]
  const float* bf2  = (const float*)d_in[12];
  const float* Wcat = (const float*)d_in[13];   // [27,256,120]
  const float* bcat = (const float*)d_in[14];
  float* out = (float*)d_out;                   // [N,128]

  const int N = in_sizes[1] / 2;                // 50000 (divisible by 16)

  // --- workspace bump allocator (256B aligned) ---
  char* ws = (char*)d_ws;
  size_t off = 0;
  auto alloc = [&](size_t bytes) -> char* {
    char* p = ws + off;
    off += (bytes + 255) & ~(size_t)255;
    return p;
  };
  __bf16* corrB = (__bf16*)alloc((size_t)N * 352 * 2);   // padded corr features
  __bf16* corr1 = (__bf16*)alloc((size_t)N * 256 * 2);
  __bf16* corr2 = (__bf16*)alloc((size_t)N * 192 * 2);
  __bf16* f1    = (__bf16*)alloc((size_t)N * 128 * 2);
  __bf16* f2    = (__bf16*)alloc((size_t)N * 64  * 2);
  __bf16* WpC1  = (__bf16*)alloc((size_t)1  * 11 * 16 * 1024);
  __bf16* WpC2  = (__bf16*)alloc((size_t)27 * 8  * 12 * 1024);
  __bf16* WpF2  = (__bf16*)alloc((size_t)27 * 4  * 4  * 1024);
  __bf16* WpCat = (__bf16*)alloc((size_t)27 * 8  * 8  * 1024);
  float*  bcatP = (float*)alloc(128 * sizeof(float));
  __bf16* cat   = corrB;   // corrB is dead after conv1; reuse for concat buffer

  // --- prep: conversions + weight packing ---
  {
    size_t tot = (size_t)N * 352;
    convert_corr_kernel<<<(unsigned)((tot + 255) / 256), 256, 0, stream>>>(corr, corrB, N);
  }
  auto pack = [&](const float* W, __bf16* Wp, int K, int Cin, int CinP, int Cout, int CoutP) {
    size_t tot = (size_t)K * (CinP >> 5) * (CoutP >> 4) * 512;
    pack_w_kernel<<<(unsigned)((tot + 255) / 256), 256, 0, stream>>>(W, Wp, K, Cin, CinP, Cout, CoutP);
  };
  pack(Wc1,  WpC1,  1,  324, 352, 256, 256);
  pack(Wc2,  WpC2,  27, 256, 256, 192, 192);
  pack(Wf2,  WpF2,  27, 128, 128, 64,  64);
  pack(Wcat, WpCat, 27, 256, 256, 120, 128);
  pad_bias_kernel<<<1, 128, 0, stream>>>(bcat, bcatP, 120, 128);

  auto conv = [&](const __bf16* feat, int CinP, const int* nbr, int K,
                  const __bf16* Wp, const float* bias,
                  __bf16* outB, float* outF, int outFcols, int CoutP) {
    int jobs = ((N + 31) / 32) * (CoutP / 64);
    wmma_subm_conv_kernel<<<(unsigned)((jobs + 7) / 8), 256, 0, stream>>>(
        feat, CinP, nbr, K, Wp, bias, outB, outF, outFcols, CoutP, N);
  };

  // conv1 (1x1, 324->256) : identity neighbor map
  conv(corrB, 352, nullptr, 1, WpC1, bc1, corr1, nullptr, 0, 256);
  // flowconv1 (k=343, 2->128) : VALU kernel
  flow_conv1_kernel<<<(unsigned)(N / 16), 128, 0, stream>>>(flow, nbr7, Wf1, bf1, f1, N);
  // conv2 (k=27, 256->192)
  conv(corr1, 256, nbr3, 27, WpC2, bc2, corr2, nullptr, 0, 192);
  // flowconv2 (k=27, 128->64)
  conv(f1, 128, nbr3, 27, WpF2, bf2, f2, nullptr, 0, 64);
  // concat [corr2 | f2] -> cat [N,256]
  {
    size_t tot = (size_t)N * 256;
    concat_kernel<<<(unsigned)((tot + 255) / 256), 256, 0, stream>>>(corr2, f2, cat, N);
  }
  // convcat (k=27, 256->120 padded 128), writes f32 directly into d_out cols 0..119
  conv(cat, 256, nbr3, 27, WpCat, bcatP, nullptr, out, 120, 128);
  // append flow + extrinsics into cols 120..127
  {
    size_t tot = (size_t)N * 8;
    tail_kernel<<<(unsigned)((tot + 255) / 256), 256, 0, stream>>>(flow, extr, out, N);
  }
}
